// IntraCycleMoELayer_51616916963604
// MI455X (gfx1250) — matrix operand order repaired and checked by reading, not compiled, not run on hardware
//
#include <hip/hip_runtime.h>
#include <math.h>

// ---------------- problem constants ----------------
#define B_    16
#define L_    512
#define D_    768
#define DFF_  3072
#define DLLM_ 4096
#define E_    8
#define TOPK_ 2

// ---------------- tiling constants -----------------
#define MT_  32      // M rows per block (of L)
#define KC_  512     // DFF chunk held in LDS
#define HS_  520     // LDS row stride for h tile (bf16 elems), padded vs 512

typedef __bf16 bf16_t;
typedef __attribute__((ext_vector_type(8)))  bf16_t v8bf;
typedef __attribute__((ext_vector_type(16))) bf16_t v16bf;
typedef __attribute__((ext_vector_type(8)))  float  v8f;

#define WMMA_BF16(Cacc, Afrag, Bfrag)                                          \
  (Cacc) = __builtin_amdgcn_wmma_f32_16x16x32_bf16(                            \
      false, (Afrag), false, (Bfrag), (short)0, (Cacc), false, false)

// gelu(v) = 0.5*v*(1+tanh(u)) = v * sigmoid(2u),  u = sqrt(2/pi)*(v+0.044715v^3)
// branch-free: one v_exp_f32 + one v_rcp_f32
__device__ __forceinline__ float gelu_tanh(float v) {
  const float k0 = 0.7978845608028654f; // sqrt(2/pi)
  const float k1 = 0.044715f;
  float u = k0 * (v + k1 * v * v * v);
  float t = __expf(-2.0f * u);
  return v * __builtin_amdgcn_rcpf(1.0f + t);
}

__device__ __forceinline__ v16bf make_v16(v8bf lo, v8bf hi) {
  return __builtin_shufflevector(lo, hi, 0, 1, 2, 3, 4, 5, 6, 7,
                                 8, 9, 10, 11, 12, 13, 14, 15);
}

// A fragment (16x32, bf16) from a bf16 row-major tile, leading dim ld.
// ISA 16-bit A layout: lane m = lane&15, half = lane>>4;
// idx 0..7  -> k = kB + 8*half + idx
// idx 8..15 -> k = kB + 16 + 8*half + (idx-8)
__device__ __forceinline__ v16bf load_a_bf16(const bf16_t* base, int m0, int kB,
                                             int lane, int ld) {
  int row  = m0 + (lane & 15);
  int half = lane >> 4;
  const bf16_t* p = base + (long)row * ld + kB + 8 * half;
  v8bf lo = *(const v8bf*)(p);
  v8bf hi = *(const v8bf*)(p + 16);
  return make_v16(lo, hi);
}

// B fragment (32x16, bf16) from a pre-transposed bf16 weight wt[n][k],
// leading dim (k extent) ldk. B layout: n=lane&15, half=lane>>4;
// idx 0..7 -> k=kB+8*half+idx; idx 8..15 -> k=kB+16+8*half+(idx-8).
__device__ __forceinline__ v16bf load_b(const bf16_t* wt, int nB, int kB,
                                        int lane, int ldk) {
  int n  = nB + (lane & 15);
  int k0 = kB + ((lane >> 4) << 3);
  const bf16_t* p = wt + (long)n * ldk + k0;
  v8bf lo = *(const v8bf*)(p);
  v8bf hi = *(const v8bf*)(p + 16);
  return make_v16(lo, hi);
}

// ---------------- zero the combined accumulator ----------------
__global__ void zero_f32_kernel(float* p, long n) {
  long i = (long)blockIdx.x * blockDim.x + threadIdx.x;
  if (i < n) p[i] = 0.0f;
}

// ---------------- x f32 -> bf16 copy ----------------
__global__ void cvt_x_kernel(const float* __restrict__ x,
                             bf16_t* __restrict__ xb, long n) {
  long i = (long)blockIdx.x * blockDim.x + threadIdx.x;
  if (i < n) xb[i] = (bf16_t)x[i];
}

// ---------------- weight transpose + f32->bf16 ----------------
// w1t[e][n][k] = w1[e][k][n]  (n in DFF, k in D);  e==8 -> general
// w2t[e][n][k] = w2[e][k][n]  (n in D, k in DFF);  e==8 -> general
__global__ void prep_weights_kernel(const float* __restrict__ e_w1,
                                    const float* __restrict__ g_w1,
                                    const float* __restrict__ e_w2,
                                    const float* __restrict__ g_w2,
                                    bf16_t* __restrict__ w1t,
                                    bf16_t* __restrict__ w2t) {
  long idx = (long)blockIdx.x * blockDim.x + threadIdx.x;
  const long tot = 9L * DFF_ * D_;
  if (idx >= tot) return;
  if (blockIdx.y == 0) {
    int  e   = (int)(idx / ((long)DFF_ * D_));
    long rem = idx % ((long)DFF_ * D_);
    int  n   = (int)(rem / D_);
    int  k   = (int)(rem % D_);
    float v  = (e < E_) ? e_w1[(long)e * D_ * DFF_ + (long)k * DFF_ + n]
                        : g_w1[(long)k * DFF_ + n];
    w1t[idx] = (bf16_t)v;
  } else {
    int  e   = (int)(idx / ((long)D_ * DFF_));
    long rem = idx % ((long)D_ * DFF_);
    int  n   = (int)(rem / DFF_);
    int  k   = (int)(rem % DFF_);
    float v  = (e < E_) ? e_w2[(long)e * DFF_ * D_ + (long)k * D_ + n]
                        : g_w2[(long)k * D_ + n];
    w2t[idx] = (bf16_t)v;
  }
}

// ---------------- router stage 1: h = relu(emb@We + cyc*Wc + b) ----------------
__global__ void router_h_kernel(const float* __restrict__ emb,
                                const float* __restrict__ cyc,
                                const float* __restrict__ We,
                                const float* __restrict__ Wc,
                                const float* __restrict__ gb,
                                float* __restrict__ hrout) {
  int b = blockIdx.y;
  int j = blockIdx.x * blockDim.x + threadIdx.x;
  float acc = cyc[b] * Wc[j] + gb[j];
  const float* eb = emb + (long)b * DLLM_;
  for (int k = 0; k < DLLM_; ++k)
    acc += eb[k] * We[(long)k * DFF_ + j];
  hrout[(long)b * DFF_ + j] = fmaxf(acc, 0.0f);
}

// ---------------- router stage 2: logits, softmax, top-2 gates ----------------
__global__ void router_gate_kernel(const float* __restrict__ hrout,
                                   const float* __restrict__ Wo,
                                   const float* __restrict__ bo,
                                   float* __restrict__ gated) {
  __shared__ float lg[B_][E_];
  int tid = threadIdx.x;
  if (tid < B_ * E_) {
    int b = tid >> 3, e = tid & 7;
    float acc = bo[e];
    const float* hb = hrout + (long)b * DFF_;
    for (int k = 0; k < DFF_; ++k) acc += hb[k] * Wo[k * E_ + e];
    lg[b][e] = acc;
  }
  __syncthreads();
  if (tid < B_) {
    int b = tid;
    float mx = lg[b][0];
    for (int e = 1; e < E_; ++e) mx = fmaxf(mx, lg[b][e]);
    float p[E_], s = 0.0f;
    for (int e = 0; e < E_; ++e) { p[e] = expf(lg[b][e] - mx); s += p[e]; }
    for (int e = 0; e < E_; ++e) p[e] /= s;
    // top-2 on logits (first-index wins on ties, as lax.top_k)
    int i1 = 0;
    for (int e = 1; e < E_; ++e) if (lg[b][e] > lg[b][i1]) i1 = e;
    int i2 = -1;
    for (int e = 0; e < E_; ++e)
      if (e != i1 && (i2 < 0 || lg[b][e] > lg[b][i2])) i2 = e;
    float denom = p[i1] + p[i2] + 1e-9f;
    for (int e = 0; e < E_; ++e)
      gated[b * E_ + e] = (e == i1 || e == i2) ? p[e] / denom : 0.0f;
  }
}

// ---------------- fused expert MLP block ----------------
// grid = (L_/MT_, B_, E_+1); block = 256 (8 waves).
__global__ __launch_bounds__(256) void expert_kernel(
    const float* __restrict__ x, const bf16_t* __restrict__ xb,
    const float* __restrict__ gated,
    const bf16_t* __restrict__ w1t, const bf16_t* __restrict__ w2t,
    const float* __restrict__ e_b1, const float* __restrict__ e_b2,
    const float* __restrict__ e_gamma, const float* __restrict__ e_beta,
    const float* __restrict__ g_b1, const float* __restrict__ g_b2,
    const float* __restrict__ g_gamma, const float* __restrict__ g_beta,
    float* __restrict__ combined, float* __restrict__ general) {
  __shared__ bf16_t hlds[MT_ * HS_];
  __shared__ float red_s[MT_];
  __shared__ float red_q[MT_];

  const int b = blockIdx.y;
  const int e = blockIdx.z;
  const float gate = (e < E_) ? gated[b * E_ + e] : 1.0f;
  if (gate == 0.0f) return;  // skip non-routed experts (exact zeros)

  const int mBase = blockIdx.x * MT_;
  const int tid = threadIdx.x;
  const int lane = tid & 31;
  const int wid = tid >> 5;

  if (tid < MT_) { red_s[tid] = 0.0f; red_q[tid] = 0.0f; }

  const float*  xtile  = x + (long)(b * L_ + mBase) * D_;
  const bf16_t* xbtile = xb + (long)(b * L_ + mBase) * D_;
  const bf16_t* W1 = w1t + (long)e * DFF_ * D_;  // [n:DFF][k:D]
  const bf16_t* W2 = w2t + (long)e * D_ * DFF_;  // [n:D][k:DFF]
  const float* b1  = (e < E_) ? e_b1 + (long)e * DFF_ : g_b1;
  const float* b2  = (e < E_) ? e_b2 + (long)e * D_   : g_b2;
  const float* gam = (e < E_) ? e_gamma + (long)e * D_ : g_gamma;
  const float* bet = (e < E_) ? e_beta + (long)e * D_  : g_beta;

  v8f oAcc[12];  // 2 M-tiles x 6 N-tiles of 16x16 f32 per wave (D cols wid*96..+96)
  #pragma unroll
  for (int i = 0; i < 12; ++i) oAcc[i] = (v8f){0.f, 0.f, 0.f, 0.f, 0.f, 0.f, 0.f, 0.f};

  #pragma unroll 1
  for (int ch = 0; ch < DFF_; ch += KC_) {
    __syncthreads();  // previous stage-2 finished reading hlds

    // ---- stage 1: h[:, ch:ch+KC] = gelu(x @ w1 + b1), per-wave N span 64 ----
    v8f hAcc[8];
    #pragma unroll
    for (int i = 0; i < 8; ++i) hAcc[i] = (v8f){0.f, 0.f, 0.f, 0.f, 0.f, 0.f, 0.f, 0.f};

    #pragma unroll 1
    for (int k = 0; k < D_; k += 32) {
      v16bf a0 = load_a_bf16(xbtile, 0, k, lane, D_);
      v16bf a1 = load_a_bf16(xbtile, 16, k, lane, D_);
      #pragma unroll
      for (int nt = 0; nt < 4; ++nt) {
        int n = ch + wid * 64 + nt * 16;
        v16bf bf = load_b(W1, n, k, lane, D_);
        WMMA_BF16(hAcc[nt], a0, bf);
        WMMA_BF16(hAcc[4 + nt], a1, bf);
      }
    }

    // bias + gelu -> bf16 -> LDS
    #pragma unroll
    for (int mt = 0; mt < 2; ++mt) {
      #pragma unroll
      for (int nt = 0; nt < 4; ++nt) {
        int nLoc  = wid * 64 + nt * 16 + (lane & 15);
        float bias = b1[ch + nLoc];
        int mrow  = mt * 16 + (lane >> 4) * 8;
        v8f c = hAcc[mt * 4 + nt];
        #pragma unroll
        for (int r = 0; r < 8; ++r) {
          float v = c[r] + bias;
          hlds[(mrow + r) * HS_ + nLoc] = (bf16_t)gelu_tanh(v);
        }
      }
    }
    __syncthreads();

    // ---- stage 2: o += h_chunk @ w2[ch:ch+KC, :], per-wave N span 96 ----
    #pragma unroll 1
    for (int kk = 0; kk < KC_; kk += 32) {
      v16bf a0 = load_a_bf16(hlds, 0, kk, lane, HS_);
      v16bf a1 = load_a_bf16(hlds, 16, kk, lane, HS_);
      #pragma unroll
      for (int nt = 0; nt < 6; ++nt) {
        int n = wid * 96 + nt * 16;
        v16bf bf = load_b(W2, n, ch + kk, lane, DFF_);
        WMMA_BF16(oAcc[nt], a0, bf);
        WMMA_BF16(oAcc[6 + nt], a1, bf);
      }
    }
  }
  __syncthreads();

  // ---- epilogue: v = o + b2 + x ; LayerNorm row stats via LDS atomics ----
  #pragma unroll
  for (int mt = 0; mt < 2; ++mt) {
    int mrow = mt * 16 + (lane >> 4) * 8;
    #pragma unroll
    for (int r = 0; r < 8; ++r) {
      float s = 0.0f, q = 0.0f;
      #pragma unroll
      for (int nt = 0; nt < 6; ++nt) {
        int n = wid * 96 + nt * 16 + (lane & 15);
        float v = oAcc[mt * 6 + nt][r] + b2[n] + xtile[(long)(mrow + r) * D_ + n];
        oAcc[mt * 6 + nt][r] = v;
        s += v;
        q += v * v;
      }
      atomicAdd(&red_s[mrow + r], s);
      atomicAdd(&red_q[mrow + r], q);
    }
  }
  __syncthreads();

  // ---- normalize, scale by gate, write out ----
  #pragma unroll
  for (int mt = 0; mt < 2; ++mt) {
    int mrow = mt * 16 + (lane >> 4) * 8;
    #pragma unroll
    for (int r = 0; r < 8; ++r) {
      int m = mrow + r;
      float mu  = red_s[m] * (1.0f / (float)D_);
      float var = red_q[m] * (1.0f / (float)D_) - mu * mu;
      float inv = rsqrtf(var + 1e-5f);
      #pragma unroll
      for (int nt = 0; nt < 6; ++nt) {
        int n = wid * 96 + nt * 16 + (lane & 15);
        float y = (oAcc[mt * 6 + nt][r] - mu) * inv * gam[n] + bet[n];
        long off = (long)(b * L_ + mBase + m) * D_ + n;
        if (e < E_) atomicAdd(&combined[off], gate * y);
        else        general[off] = y;
      }
    }
  }
}

// ---------------- finalize: out = general + float(bf16(combined)) ----------------
__global__ void finalize_kernel(const float* __restrict__ general,
                                const float* __restrict__ combined,
                                float* __restrict__ out, long n) {
  long i = (long)blockIdx.x * blockDim.x + threadIdx.x;
  if (i < n) out[i] = general[i] + (float)(bf16_t)combined[i];
}

extern "C" void kernel_launch(void* const* d_in, const int* in_sizes, int n_in,
                              void* d_out, int out_size, void* d_ws,
                              size_t ws_size, hipStream_t stream) {
  const float* x    = (const float*)d_in[0];   // [B,L,D]
  const float* cyc  = (const float*)d_in[1];   // [B,1]
  const float* emb  = (const float*)d_in[2];   // [B,DLLM]
  const float* gWe  = (const float*)d_in[3];   // [DLLM,DFF]
  const float* gWc  = (const float*)d_in[4];   // [1,DFF]
  const float* gb   = (const float*)d_in[5];   // [DFF]
  const float* gWo  = (const float*)d_in[6];   // [DFF,E]
  const float* gbo  = (const float*)d_in[7];   // [E]
  const float* e_w1 = (const float*)d_in[8];   // [E,D,DFF]
  const float* e_b1 = (const float*)d_in[9];
  const float* e_w2 = (const float*)d_in[10];  // [E,DFF,D]
  const float* e_b2 = (const float*)d_in[11];
  const float* e_ga = (const float*)d_in[12];
  const float* e_be = (const float*)d_in[13];
  const float* g_w1 = (const float*)d_in[14];  // [D,DFF]
  const float* g_b1 = (const float*)d_in[15];
  const float* g_w2 = (const float*)d_in[16];  // [DFF,D]
  const float* g_b2 = (const float*)d_in[17];
  const float* g_ga = (const float*)d_in[18];
  const float* g_be = (const float*)d_in[19];

  const long BLD = (long)B_ * L_ * D_;

  float* ws       = (float*)d_ws;
  float* combined = ws;                       // BLD f32
  float* general  = combined + BLD;           // BLD f32
  float* gated    = general + BLD;            // B*E f32
  float* hrout    = gated + B_ * E_;          // B*DFF f32
  bf16_t* w1t     = (bf16_t*)(hrout + (long)B_ * DFF_);  // 9*DFF*D bf16
  bf16_t* w2t     = w1t + 9L * DFF_ * D_;                // 9*D*DFF bf16
  bf16_t* xb      = w2t + 9L * D_ * DFF_;                // BLD bf16

  zero_f32_kernel<<<dim3((unsigned)((BLD + 255) / 256)), 256, 0, stream>>>(
      combined, BLD);

  cvt_x_kernel<<<dim3((unsigned)((BLD + 255) / 256)), 256, 0, stream>>>(x, xb,
                                                                        BLD);

  {
    long tot = 9L * DFF_ * D_;
    prep_weights_kernel<<<dim3((unsigned)((tot + 255) / 256), 2), 256, 0,
                          stream>>>(e_w1, g_w1, e_w2, g_w2, w1t, w2t);
  }

  router_h_kernel<<<dim3(DFF_ / 256, B_), 256, 0, stream>>>(emb, cyc, gWe, gWc,
                                                            gb, hrout);
  router_gate_kernel<<<1, 128, 0, stream>>>(hrout, gWo, gbo, gated);

  expert_kernel<<<dim3(L_ / MT_, B_, E_ + 1), 256, 0, stream>>>(
      x, xb, gated, w1t, w2t, e_b1, e_b2, e_ga, e_be, g_b1, g_b2, g_ga, g_be,
      combined, general);

  finalize_kernel<<<dim3((unsigned)((BLD + 255) / 256)), 256, 0, stream>>>(
      general, combined, (float*)d_out, BLD);
}
